// SequenceGRU_13537736917204
// MI455X (gfx1250) — matrix operand order repaired and checked by reading, compiled, not run on hardware
//
#include <hip/hip_runtime.h>

typedef __bf16 bf16;
typedef __attribute__((ext_vector_type(16))) __bf16 v16bf;
typedef __attribute__((ext_vector_type(8)))  __bf16 v8bf;
typedef __attribute__((ext_vector_type(8)))  float  v8f;

#define B_   256
#define T_   2048
#define D_   64
#define H_   256
#define G_   768      // 3*H
#define C_   32
#define HEAD_ 32
#define NWG  16
#define WHH_LD 264    // padded K-stride (elements) -> conflict-free LDS b128
#define WIH_LD 72

#define WMMA_BF16(a, b, c) \
  __builtin_amdgcn_wmma_f32_16x16x32_bf16(false, (a), false, (b), (short)0, (c), false, false)

__device__ __forceinline__ v16bf cat8(v8bf lo, v8bf hi) {
  return __builtin_shufflevector(lo, hi, 0,1,2,3,4,5,6,7,8,9,10,11,12,13,14,15);
}

// A fragment (16x32 bf16, ISA layout): lane<16 -> row M=lane, K {0..7,16..23};
// lane>=16 -> row M=lane-16, K {8..15,24..31}. Two contiguous 16B loads.
__device__ __forceinline__ v16bf load_a(const bf16* __restrict__ base, int stride,
                                        int m0, int k0, int lane) {
  const bf16* p = base + (size_t)(m0 + (lane & 15)) * stride + k0 + ((lane >> 4) << 3);
  v8bf lo = *(const v8bf*)p;
  v8bf hi = *(const v8bf*)(p + 16);
  return cat8(lo, hi);
}

// B fragment (32x16 bf16): lane holds column N=lane%16, K {0..7,16..23} (lane<16) / {8..15,24..31} (lane>=16).
// Source is column-major-by-N slice in LDS (== native w_hh row-major rows).
__device__ __forceinline__ v16bf load_b(const bf16* __restrict__ base, int ld,
                                        int col0, int k0, int lane) {
  const bf16* p = base + (size_t)(col0 + (lane & 15)) * ld + k0 + ((lane >> 4) << 3);
  v8bf lo = *(const v8bf*)p;
  v8bf hi = *(const v8bf*)(p + 16);
  return cat8(lo, hi);
}

__device__ __forceinline__ void grid_barrier(int* bar, int phase) {
  __threadfence();
  __syncthreads();
  if (threadIdx.x == 0) {
    __hip_atomic_fetch_add(bar, 1, __ATOMIC_RELEASE, __HIP_MEMORY_SCOPE_AGENT);
    const int target = phase * NWG;
    while (__hip_atomic_load(bar, __ATOMIC_ACQUIRE, __HIP_MEMORY_SCOPE_AGENT) < target)
      __builtin_amdgcn_s_sleep(2);
  }
  __syncthreads();
  __threadfence();
}

// ---------------- persistent GRU kernel: 16 WGs x 256 threads ----------------
__global__ __launch_bounds__(256)
void gru_persistent(const bf16* __restrict__ seq_bf,   // [T][B][D] bf16 (time-major)
                    const bf16* __restrict__ whh_bf,   // [3H][H] bf16 (native layout == B-matrix)
                    const bf16* __restrict__ wih_bf,   // [3H][D] bf16
                    const float* __restrict__ b_ih, const float* __restrict__ b_hh,
                    const float* __restrict__ context,
                    const float* __restrict__ w1, const float* __restrict__ b1,
                    const float* __restrict__ w2, const float* __restrict__ b2,
                    bf16* __restrict__ hbf0, bf16* __restrict__ hbf1,
                    float* __restrict__ h_last,
                    int* __restrict__ bar,
                    float* __restrict__ out) {
  __shared__ bf16  lds_whh[48 * WHH_LD];   // 48 gate-cols x K=256 (padded)
  __shared__ bf16  lds_wih[48 * WIH_LD];   // 48 gate-cols x K=64  (padded)
  __shared__ float h_state[B_ * 16];       // f32 state for this WG's 16 hidden cols

  const int wg   = blockIdx.x;
  const int tid  = threadIdx.x;
  const int lane = tid & 31;
  const int wave = tid >> 5;
  const int j0   = wg * 16;                // this WG's hidden columns [j0, j0+16)
  const int jl   = lane & 15;

  // ---- stage weight slices into LDS (once; compiler promotes the 24 B-frags
  //      to persistent VGPRs via s_set_vgpr_msb, reused for all 2048 steps) ----
  for (int e = tid; e < 48 * 32; e += 256) {           // w_hh: 48 cols x 32 chunks of 8
    int col = e >> 5, ch = e & 31;
    int grow = (col >> 4) * H_ + j0 + (col & 15);      // gate*H + j
    *(v8bf*)&lds_whh[col * WHH_LD + ch * 8] = *(const v8bf*)&whh_bf[(size_t)grow * H_ + ch * 8];
  }
  for (int e = tid; e < 48 * 8; e += 256) {            // w_ih: 48 cols x 8 chunks of 8
    int col = e >> 3, ch = e & 7;
    int grow = (col >> 4) * H_ + j0 + (col & 15);
    *(v8bf*)&lds_wih[col * WIH_LD + ch * 8] = *(const v8bf*)&wih_bf[(size_t)grow * D_ + ch * 8];
  }
  for (int e = tid; e < B_ * 16; e += 256) h_state[e] = 0.0f;
  for (int e = tid; e < (B_ * H_) / NWG; e += 256)     // zero t=0 read buffer slice
    hbf0[(size_t)wg * (B_ * H_ / NWG) + e] = (bf16)0.0f;

  // per-lane gate biases (column = j0+jl for every acc element in this lane)
  const float bias_r  = b_ih[j0 + jl]          + b_hh[j0 + jl];
  const float bias_z  = b_ih[H_ + j0 + jl]     + b_hh[H_ + j0 + jl];
  const float bias_nx = b_ih[2 * H_ + j0 + jl];
  const float bias_nh = b_hh[2 * H_ + j0 + jl];

  grid_barrier(bar, 1);

  // ---- recurrence ----
  for (int t = 0; t < T_; ++t) {
    const bf16* hr = (t & 1) ? hbf1 : hbf0;    // read h_{t}
    bf16*       hw = (t & 1) ? hbf0 : hbf1;    // write h_{t+1}
    const bf16* xs = seq_bf + (size_t)t * (B_ * D_);
    if (t + 1 < T_ && lane == 0)
      __builtin_prefetch(seq_bf + (size_t)(t + 1) * (B_ * D_), 0, 1);

    // Two independent M-tiles per wave, fully unrolled: tile-1's A-fragment
    // L2 loads overlap tile-0's WMMAs (only 2 waves/SIMD -> need the ILP).
    #pragma unroll
    for (int mi = 0; mi < 2; ++mi) {
      const int m0 = (wave * 2 + mi) * 16;     // batch tile
      v8f ar = {}, az = {}, anx = {}, anh = {};

      // fused input projection: K = 64 over w_ih
      #pragma unroll
      for (int kb = 0; kb < 2; ++kb) {
        v16bf a  = load_a(xs, D_, m0, kb * 32, lane);
        v16bf br = load_b(lds_wih, WIH_LD, 0,  kb * 32, lane);
        v16bf bz = load_b(lds_wih, WIH_LD, 16, kb * 32, lane);
        v16bf bn = load_b(lds_wih, WIH_LD, 32, kb * 32, lane);
        ar  = WMMA_BF16(a, br, ar);
        az  = WMMA_BF16(a, bz, az);
        anx = WMMA_BF16(a, bn, anx);
      }
      // hidden projection: K = 256 over w_hh, A straight from L2-resident h broadcast
      #pragma unroll
      for (int kb = 0; kb < 8; ++kb) {
        v16bf a  = load_a(hr, H_, m0, kb * 32, lane);
        v16bf br = load_b(lds_whh, WHH_LD, 0,  kb * 32, lane);
        v16bf bz = load_b(lds_whh, WHH_LD, 16, kb * 32, lane);
        v16bf bn = load_b(lds_whh, WHH_LD, 32, kb * 32, lane);
        ar  = WMMA_BF16(a, br, ar);
        az  = WMMA_BF16(a, bz, az);
        anh = WMMA_BF16(a, bn, anh);
      }

      // elementwise GRU update; C-layout: elem v -> (batch m0+(lane/16)*8+v, col jl)
      const int mrow = m0 + ((lane >> 4) << 3);
      #pragma unroll
      for (int v = 0; v < 8; ++v) {
        float r  = 1.0f / (1.0f + __expf(-(ar[v] + bias_r)));
        float z  = 1.0f / (1.0f + __expf(-(az[v] + bias_z)));
        float n  = tanhf(anx[v] + bias_nx + r * (anh[v] + bias_nh));
        float ho = h_state[(mrow + v) * 16 + jl];
        float hn = (1.0f - z) * n + z * ho;
        h_state[(mrow + v) * 16 + jl] = hn;
        hw[(size_t)(mrow + v) * H_ + j0 + jl] = (bf16)hn;
      }
    }
    grid_barrier(bar, t + 2);
  }

  // ---- publish final hidden state ----
  for (int e = tid; e < B_ * 16; e += 256) {
    int b = e >> 4, c = e & 15;
    h_last[(size_t)b * H_ + j0 + c] = h_state[e];
  }
  grid_barrier(bar, T_ + 2);

  // ---- tiny MLP head (2.4 MFLOP): WG 0, one batch row per thread ----
  if (wg == 0) {
    const int b = tid;
    float o0 = b2[0], o1 = b2[1];
    for (int hd = 0; hd < HEAD_; ++hd) {
      const float* w1r = w1 + hd * (H_ + C_);
      float s = b1[hd];
      for (int i = 0; i < H_; ++i) s += h_last[(size_t)b * H_ + i] * w1r[i];
      for (int i = 0; i < C_; ++i) s += context[b * C_ + i] * w1r[H_ + i];
      s = fmaxf(s, 0.0f);
      o0 += s * w2[hd];
      o1 += s * w2[HEAD_ + hd];
    }
    out[b * 2 + 0] = o0;
    out[b * 2 + 1] = o1;
  }
}

// ---------------- prep kernels ----------------
__global__ void init_ws(int* bar) {
  if (threadIdx.x == 0 && blockIdx.x == 0) *bar = 0;
}

__global__ void cvt_weights(const float* __restrict__ whh, const float* __restrict__ wih,
                            bf16* __restrict__ whh_bf, bf16* __restrict__ wih_bf) {
  int i = blockIdx.x * 256 + threadIdx.x;
  if (i < G_ * H_) whh_bf[i] = (bf16)whh[i];
  if (i < G_ * D_) wih_bf[i] = (bf16)wih[i];
}

// seq [B][T][D] f32  ->  seq_bf [T][B][D] bf16 (time-major, contiguous per step)
__global__ void cvt_seq(const float* __restrict__ seq, bf16* __restrict__ seq_bf) {
  int t = blockIdx.x >> 8;
  int b = blockIdx.x & 255;
  int d = threadIdx.x;
  seq_bf[((size_t)t * B_ + b) * D_ + d] = (bf16)seq[((size_t)b * T_ + t) * D_ + d];
}

extern "C" void kernel_launch(void* const* d_in, const int* in_sizes, int n_in,
                              void* d_out, int out_size, void* d_ws, size_t ws_size,
                              hipStream_t stream) {
  const float* seq     = (const float*)d_in[0];
  const float* context = (const float*)d_in[1];
  const float* w_ih    = (const float*)d_in[2];
  const float* w_hh    = (const float*)d_in[3];
  const float* b_ih    = (const float*)d_in[4];
  const float* b_hh    = (const float*)d_in[5];
  const float* w1      = (const float*)d_in[6];
  const float* b1      = (const float*)d_in[7];
  const float* w2      = (const float*)d_in[8];
  const float* b2      = (const float*)d_in[9];

  char* ws = (char*)d_ws;
  size_t off = 0;
  int*   bar     = (int*)(ws + off);  off += 256;
  bf16*  hbf0    = (bf16*)(ws + off); off += (size_t)B_ * H_ * 2;
  bf16*  hbf1    = (bf16*)(ws + off); off += (size_t)B_ * H_ * 2;
  float* h_last  = (float*)(ws + off); off += (size_t)B_ * H_ * 4;
  bf16*  whh_bf  = (bf16*)(ws + off); off += (size_t)G_ * H_ * 2;
  bf16*  wih_bf  = (bf16*)(ws + off); off += (size_t)G_ * D_ * 2;
  bf16*  seq_bf  = (bf16*)(ws + off); off += (size_t)T_ * B_ * D_ * 2;  // ~67 MB

  init_ws<<<1, 64, 0, stream>>>(bar);
  cvt_weights<<<(G_ * H_ + 255) / 256, 256, 0, stream>>>(w_hh, w_ih, whh_bf, wih_bf);
  cvt_seq<<<T_ * B_, D_, 0, stream>>>(seq, seq_bf);
  gru_persistent<<<NWG, 256, 0, stream>>>(seq_bf, whh_bf, wih_bf, b_ih, b_hh,
                                          context, w1, b1, w2, b2,
                                          hbf0, hbf1, h_last, bar, (float*)d_out);
}
